// RQVAE_72292889526321
// MI455X (gfx1250) — compile-verified, matrix operand order checked by reading, not run on hardware
//
#include <hip/hip_runtime.h>
#include <hip/hip_bf16.h>
#include <cstdint>

// ---------------------------------------------------------------------------
// RQ-VAE forward on gfx1250:
//  - bf16 WMMA GEMMs (v_wmma_f32_16x16x32_bf16), f32 accumulate
//  - weights pre-transposed to [N][K] so A and B tiles stage identically as
//    row-major [128][32] 16-byte chunks
//  - double-buffered LDS staging via GLOBAL_LOAD_ASYNC_TO_LDS_B128 (ASYNCcnt)
//  - fp32 residual quantization with wave32 shuffle argmin
// ---------------------------------------------------------------------------

typedef __attribute__((ext_vector_type(16))) __bf16 v16bf;
typedef __attribute__((ext_vector_type(8)))  float  v8f;

union FragBF { v16bf v; uint32_t u[8]; };
union AccF   { v8f   v; float    f[8]; };

static __device__ __forceinline__ __bf16 f2bf(float f) {
  union { float f; uint32_t u; } a; a.f = f;
  uint32_t r = a.u + 0x7FFFu + ((a.u >> 16) & 1u);   // round-to-nearest-even
  union { uint16_t u; __bf16 h; } b; b.u = (uint16_t)(r >> 16);
  return b.h;
}

#if defined(__gfx1250__) && __has_builtin(__builtin_amdgcn_global_load_async_to_lds_b128)
#define HAVE_ASYNC_LDS 1
typedef int i32x4 __attribute__((vector_size(16)));
typedef __attribute__((address_space(1))) i32x4 gas_i32x4;   // global
typedef __attribute__((address_space(3))) i32x4 las_i32x4;   // LDS
#else
#define HAVE_ASYNC_LDS 0
#endif

// ------------------------------ constants ----------------------------------
#define BATCH   32768
#define BM      128
#define BN      128
#define BK      32
#define LDT     40      // padded LDS row stride (bf16 elems), multiple of 8

// ------------------------- f32 -> bf16 convert -----------------------------
__global__ void conv_f32_bf16(const float* __restrict__ in,
                              __bf16* __restrict__ out, int n) {
  int i = blockIdx.x * blockDim.x + threadIdx.x;
  if (i < n) out[i] = f2bf(in[i]);
}

// ------------- weight convert + transpose: W[K,N] -> Wt[N,K] bf16 ----------
__global__ __launch_bounds__(256)
void transpose_f32_bf16(const float* __restrict__ in,  // [K,N]
                        __bf16* __restrict__ out,      // [N,K]
                        int K, int N) {
  int k = blockIdx.x * 32 + (threadIdx.x & 31);
  int n = blockIdx.y * 8  + (threadIdx.x >> 5);
  if (k < K && n < N) out[(size_t)n * K + k] = f2bf(in[(size_t)k * N + n]);
}

// ------------- codebook prep: bf16 copy (already [K=1024][D=128]) + ||c||^2
__global__ __launch_bounds__(128)
void cb_prep(const float* __restrict__ cb,      // [1024,128] f32
             __bf16* __restrict__ cbB,          // [1024,128] bf16 (== Wt form)
             float* __restrict__ cbsq) {        // [1024] f32
  __shared__ float red[128];
  int k = blockIdx.x;
  int d = threadIdx.x;
  float v = cb[(size_t)k * 128 + d];
  cbB[(size_t)k * 128 + d] = f2bf(v);
  red[d] = v * v;
  __syncthreads();
  for (int s = 64; s > 0; s >>= 1) {
    if (d < s) red[d] += red[d + s];
    __syncthreads();
  }
  if (d == 0) cbsq[k] = red[0];
}

// ------------------------- bf16 WMMA GEMM ----------------------------------
// C[M,N] = act( alpha * (A[M,K] @ Wt[N,K]^T) + bias[N] )
// A, Wt bf16 row-major; accum f32. Requires M%128==0, N%128==0, K%32==0.
template<int RELU, int WF, int WB>
__global__ __launch_bounds__(256)
void gemm_bf16_wmma(const __bf16* __restrict__ A, const __bf16* __restrict__ Wt,
                    const float* __restrict__ bias, float alpha,
                    __bf16* __restrict__ outb, float* __restrict__ outf,
                    int M, int N, int K) {
  __shared__ __bf16 As[2][BM * LDT];
  __shared__ __bf16 Bs[2][BM * LDT];

  const int tid  = threadIdx.x;
  const int lane = tid & 31;
  const int wid  = tid >> 5;        // 8 waves
  const int wm   = wid >> 2;        // 0..1  -> 64-row strip
  const int wn   = wid & 3;         // 0..3  -> 32-col strip
  const int bm   = blockIdx.y * BM;
  const int bn   = blockIdx.x * BN;
  const int hl   = lane >> 4;       // wave half
  const int l15  = lane & 15;

  // stage one 128x32 tile of A and of Wt (both row-major, K-contiguous)
  auto stage = [&](int s, int kb) {
#pragma unroll
    for (int p = 0; p < 2; ++p) {
      int c = p * 256 + tid;
      int r = c >> 2, col = (c & 3) * 8;
      const __bf16* ga = A  + (size_t)(bm + r) * K + kb + col;
      const __bf16* gb = Wt + (size_t)(bn + r) * K + kb + col;
      __bf16* la = &As[s][r * LDT + col];
      __bf16* lb = &Bs[s][r * LDT + col];
#if HAVE_ASYNC_LDS
      __builtin_amdgcn_global_load_async_to_lds_b128(
          (gas_i32x4*)ga, (las_i32x4*)la, 0, 0);
      __builtin_amdgcn_global_load_async_to_lds_b128(
          (gas_i32x4*)gb, (las_i32x4*)lb, 0, 0);
#else
      *(uint4*)la = *(const uint4*)ga;
      *(uint4*)lb = *(const uint4*)gb;
#endif
    }
  };
  auto flush = [&]() {
#if HAVE_ASYNC_LDS
#if __has_builtin(__builtin_amdgcn_s_wait_asynccnt)
    __builtin_amdgcn_s_wait_asynccnt(0);
#endif
#endif
    __syncthreads();
  };

  v8f acc[4][2];
  const v8f zacc = {0.f,0.f,0.f,0.f,0.f,0.f,0.f,0.f};
#pragma unroll
  for (int mi = 0; mi < 4; ++mi)
#pragma unroll
    for (int ni = 0; ni < 2; ++ni) acc[mi][ni] = zacc;

  const int nk = K / BK;
  stage(0, 0);
  flush();

  for (int t = 0; t < nk; ++t) {
    const int cur = t & 1;
    if (t + 1 < nk) stage(cur ^ 1, (t + 1) * BK);   // overlap copy w/ compute

    // fragments per the documented 16-bit A/B VGPR layout:
    // lane holds row (A) / col (B); VGPR pair p holds consecutive K
    FragBF afr[4], bfr[2];
#pragma unroll
    for (int mi = 0; mi < 4; ++mi) {
      const __bf16* ar = &As[cur][(wm * 64 + mi * 16 + l15) * LDT];
#pragma unroll
      for (int p = 0; p < 8; ++p) {
        int kk = (p < 4) ? (hl * 8 + 2 * p) : (16 + hl * 8 + 2 * (p - 4));
        afr[mi].u[p] = *(const uint32_t*)(ar + kk);
      }
    }
#pragma unroll
    for (int ni = 0; ni < 2; ++ni) {
      const __bf16* br = &Bs[cur][(wn * 32 + ni * 16 + l15) * LDT];
#pragma unroll
      for (int p = 0; p < 8; ++p) {
        int kk = (p < 4) ? (hl * 8 + 2 * p) : (16 + hl * 8 + 2 * (p - 4));
        bfr[ni].u[p] = *(const uint32_t*)(br + kk);
      }
    }
#pragma unroll
    for (int mi = 0; mi < 4; ++mi)
#pragma unroll
      for (int ni = 0; ni < 2; ++ni)
        acc[mi][ni] = __builtin_amdgcn_wmma_f32_16x16x32_bf16(
            false, afr[mi].v, false, bfr[ni].v,
            (short)0, acc[mi][ni], false, false);
    flush();
  }

  // ---- epilogue: alpha*acc + bias, optional relu, compile-time store mix ----
#pragma unroll
  for (int mi = 0; mi < 4; ++mi) {
#pragma unroll
    for (int ni = 0; ni < 2; ++ni) {
      int col  = bn + wn * 32 + ni * 16 + l15;
      float bv = bias[col];
      int row0 = bm + wm * 64 + mi * 16 + hl * 8;
      size_t base = (size_t)row0 * N + col;
      AccF au; au.v = acc[mi][ni];
#pragma unroll
      for (int v = 0; v < 8; ++v) {
        float val = alpha * au.f[v] + bv;
        if (RELU) val = fmaxf(val, 0.0f);
        size_t idx = base + (size_t)v * N;
        if (WF) outf[idx] = val;
        if (WB) outb[idx] = f2bf(val);
      }
    }
  }
}

// --------------------- residual quantization step --------------------------
__global__ __launch_bounds__(256)
void quantize_step(const float* __restrict__ scores,  // [B,1024] = ||c||^2 - 2 r.c
                   const float* __restrict__ z,       // [B,128] f32
                   const float* __restrict__ cb,      // [1024,128] f32
                   float* __restrict__ zhat,          // [B,128] f32
                   __bf16* __restrict__ resb,         // [B,128] bf16
                   __bf16* __restrict__ zhatb,        // [B,128] bf16 (last round)
                   float* __restrict__ codes,         // [B,3] (as float)
                   int depth, int first, int last) {
  int wid = threadIdx.x >> 5, lane = threadIdx.x & 31;
  int b = blockIdx.x * 8 + wid;
  const float* srow = scores + (size_t)b * 1024;

  float bv = 3.4e38f; int bi = 0;
  for (int k = lane; k < 1024; k += 32) {
    float v = srow[k];
    if (v < bv) { bv = v; bi = k; }
  }
  for (int off = 16; off > 0; off >>= 1) {
    float ov = __shfl_xor(bv, off, 32);
    int   oi = __shfl_xor(bi, off, 32);
    if (ov < bv || (ov == bv && oi < bi)) { bv = ov; bi = oi; }
  }

  const float* crow = cb + (size_t)bi * 128;
#pragma unroll
  for (int j = 0; j < 4; ++j) {
    int dd = lane * 4 + j;
    size_t idx = (size_t)b * 128 + dd;
    float prev = first ? 0.0f : zhat[idx];
    float zh = prev + crow[dd];
    zhat[idx] = zh;
    resb[idx] = f2bf(z[idx] - zh);
    if (last) zhatb[idx] = f2bf(zh);
  }
  if (lane == 0) codes[(size_t)b * 3 + depth] = (float)bi;
}

// ----------------------------- host driver ---------------------------------
extern "C" void kernel_launch(void* const* d_in, const int* in_sizes, int n_in,
                              void* d_out, int out_size, void* d_ws, size_t ws_size,
                              hipStream_t stream) {
  (void)in_sizes; (void)n_in; (void)out_size; (void)ws_size;

  const int B = BATCH;
  const float* x  = (const float*)d_in[0];
  const float* eW[4] = { (const float*)d_in[1], (const float*)d_in[3],
                         (const float*)d_in[5], (const float*)d_in[7] };
  const float* eB[4] = { (const float*)d_in[2], (const float*)d_in[4],
                         (const float*)d_in[6], (const float*)d_in[8] };
  const float* dW[4] = { (const float*)d_in[9],  (const float*)d_in[11],
                         (const float*)d_in[13], (const float*)d_in[15] };
  const float* dB[4] = { (const float*)d_in[10], (const float*)d_in[12],
                         (const float*)d_in[14], (const float*)d_in[16] };
  const float* cb = (const float*)d_in[17];
  float* out = (float*)d_out;
  float* codes = out + (size_t)B * 768;

  static const int encD[5] = { 768, 2048, 1024, 512, 128 };
  static const int decD[5] = { 128, 2048, 1024, 512, 768 };

  // -------- workspace layout --------
  char* ws = (char*)d_ws;
  size_t o = 0;
  auto take = [&](size_t bytes) -> void* {
    o = (o + 255) & ~(size_t)255;
    void* p = ws + o; o += bytes; return p;
  };
  __bf16* xb = (__bf16*)take((size_t)B * 768 * 2);
  __bf16* eWt[4]; __bf16* dWt[4];                     // transposed [N][K] bf16
  for (int i = 0; i < 4; ++i)
    eWt[i] = (__bf16*)take((size_t)encD[i] * encD[i + 1] * 2);
  for (int i = 0; i < 4; ++i)
    dWt[i] = (__bf16*)take((size_t)decD[i] * decD[i + 1] * 2);
  __bf16* cbB   = (__bf16*)take((size_t)1024 * 128 * 2);
  float*  cbsq  = (float*) take((size_t)1024 * 4);
  void*   bufA  = take((size_t)B * 2048 * 2);   // aliases scores [B,1024] f32
  void*   bufB  = take((size_t)B * 1024 * 2);
  float*  z     = (float*) take((size_t)B * 128 * 4);
  float*  zhat  = (float*) take((size_t)B * 128 * 4);
  __bf16* resb  = (__bf16*)take((size_t)B * 128 * 2);
  __bf16* zhatb = (__bf16*)take((size_t)B * 128 * 2);
  float*  scores = (float*)bufA;

  // -------- one-time conversions --------
  conv_f32_bf16<<<(unsigned)(((size_t)B * 768 + 255) / 256), 256, 0, stream>>>(
      x, xb, B * 768);
  auto wtr = [&](const float* src, __bf16* dst, int K, int N) {
    dim3 g((K + 31) / 32, (N + 7) / 8);
    transpose_f32_bf16<<<g, 256, 0, stream>>>(src, dst, K, N);
  };
  for (int i = 0; i < 4; ++i) wtr(eW[i], eWt[i], encD[i], encD[i + 1]);
  for (int i = 0; i < 4; ++i) wtr(dW[i], dWt[i], decD[i], decD[i + 1]);
  cb_prep<<<1024, 128, 0, stream>>>(cb, cbB, cbsq);

  auto gemm = [&](const __bf16* A, const __bf16* Wt, const float* bias,
                  float alpha, int relu, __bf16* ob, float* of,
                  int M, int N, int K) {
    dim3 g(N / BN, M / BM);
    if (relu)
      gemm_bf16_wmma<1,0,1><<<g, 256, 0, stream>>>(A, Wt, bias, alpha, ob, of, M, N, K);
    else if (ob && of)
      gemm_bf16_wmma<0,1,1><<<g, 256, 0, stream>>>(A, Wt, bias, alpha, ob, of, M, N, K);
    else if (of)
      gemm_bf16_wmma<0,1,0><<<g, 256, 0, stream>>>(A, Wt, bias, alpha, ob, of, M, N, K);
    else
      gemm_bf16_wmma<0,0,1><<<g, 256, 0, stream>>>(A, Wt, bias, alpha, ob, of, M, N, K);
  };

  // -------- encoder --------
  __bf16* a1 = (__bf16*)bufA;
  __bf16* a2 = (__bf16*)bufB;
  __bf16* a3 = (__bf16*)bufA;
  gemm(xb, eWt[0], eB[0], 1.0f, 1, a1, nullptr, B, encD[1], encD[0]); // 768->2048
  gemm(a1, eWt[1], eB[1], 1.0f, 1, a2, nullptr, B, encD[2], encD[1]); // 2048->1024
  gemm(a2, eWt[2], eB[2], 1.0f, 1, a3, nullptr, B, encD[3], encD[2]); // 1024->512
  // z (f32) + initial residue bf16 (z_hat == 0 so residue == z)
  gemm(a3, eWt[3], eB[3], 1.0f, 0, resb, z, B, encD[4], encD[3]);     // 512->128

  // -------- residual quantization (DEPTH = 3) --------
  for (int d = 0; d < 3; ++d) {
    // scores = ||c||^2 - 2 * residue @ codebook^T  (cbB[k][d] is Wt layout)
    gemm(resb, cbB, cbsq, -2.0f, 0, nullptr, scores, B, 1024, 128);
    quantize_step<<<B / 8, 256, 0, stream>>>(scores, z, cb, zhat, resb, zhatb,
                                             codes, d, d == 0 ? 1 : 0,
                                             d == 2 ? 1 : 0);
  }

  // -------- decoder (forward value of straight-through == z_hat) --------
  __bf16* b1 = (__bf16*)bufA;
  __bf16* b2 = (__bf16*)bufB;
  __bf16* b3 = (__bf16*)bufA;
  gemm(zhatb, dWt[0], dB[0], 1.0f, 1, b1, nullptr, B, decD[1], decD[0]); // 128->2048
  gemm(b1,    dWt[1], dB[1], 1.0f, 1, b2, nullptr, B, decD[2], decD[1]); // 2048->1024
  gemm(b2,    dWt[2], dB[2], 1.0f, 1, b3, nullptr, B, decD[3], decD[2]); // 1024->512
  gemm(b3,    dWt[3], dB[3], 1.0f, 0, nullptr, out, B, decD[4], decD[3]); // 512->768
}
